// DuoAttention_31327491457607
// MI455X (gfx1250) — matrix operand order repaired and checked by reading, not compiled, not run on hardware
//
#include <hip/hip_runtime.h>
#include <hip/hip_bf16.h>

// ---------------------------------------------------------------------------
// DuoAttention for MI455X (gfx1250, wave32, WMMA).
// WINDOW == S  =>  local == full  =>  out[i,j,s,d] = f(m[i]) * full[j,s,d],
// f(m) = (m < 0.1) ? (1-m) : 1.  One MHA in bf16 WMMA + gating.
// ---------------------------------------------------------------------------

typedef __attribute__((ext_vector_type(16))) __bf16 v16bf;
typedef __attribute__((ext_vector_type(8)))  __bf16 v8bf;
typedef __attribute__((ext_vector_type(8)))  float  v8f;
typedef __attribute__((ext_vector_type(4)))  int    v4i;

static constexpr int BB = 2;
static constexpr int SS = 2048;
static constexpr int DD = 1024;
static constexpr int HH = 16;
static constexpr int DH = 64;      // head dim
static constexpr int KT = 32;      // key tile
static constexpr int QT = 128;     // q rows per block (8 waves * 16)

#define WMMA_BF16(a, b, c) \
  __builtin_amdgcn_wmma_f32_16x16x32_bf16(false, (a), false, (b), (short)0, (c), false, false)

#if __has_builtin(__builtin_amdgcn_global_load_async_to_lds_b128) && \
    __has_builtin(__builtin_amdgcn_s_wait_asynccnt)
#define HAVE_ASYNC_LDS 1
#else
#define HAVE_ASYNC_LDS 0
#endif

// 16-byte direct-to-LDS async copy (falls back to reg staging if unavailable).
__device__ __forceinline__ void copy_b128_to_lds(const __bf16* g, __bf16* l) {
#if HAVE_ASYNC_LDS
  __builtin_amdgcn_global_load_async_to_lds_b128((v4i*)g, (v4i*)l,
                                                 /*offset=*/0, /*cpol=*/0);
#else
  *(v8bf*)l = *(const v8bf*)g;
#endif
}
__device__ __forceinline__ void async_wait_upto(int n) {
#if HAVE_ASYNC_LDS
  switch (n) {   // builtin wants an immediate
    case 0: __builtin_amdgcn_s_wait_asynccnt(0); break;
    case 1: __builtin_amdgcn_s_wait_asynccnt(1); break;
    default: __builtin_amdgcn_s_wait_asynccnt(2); break;
  }
#endif
}

// two aligned 16B LDS/global loads -> one 16-element bf16 WMMA operand
__device__ __forceinline__ v16bf load_bf16x16(const __bf16* p) {
  v8bf lo = *(const v8bf*)p;
  v8bf hi = *(const v8bf*)(p + 8);
  return __builtin_shufflevector(lo, hi, 0, 1, 2, 3, 4, 5, 6, 7,
                                 8, 9, 10, 11, 12, 13, 14, 15);
}

// max across the 16-lane half-wave group (xor masks 1,2,4,8 stay in-group)
__device__ __forceinline__ float group_max(float x) {
  x = fmaxf(x, __shfl_xor(x, 1));
  x = fmaxf(x, __shfl_xor(x, 2));
  x = fmaxf(x, __shfl_xor(x, 4));
  x = fmaxf(x, __shfl_xor(x, 8));
  return x;
}

// ------------------------------ fp32 -> bf16 -------------------------------
__global__ void cvt_kernel(const float* __restrict__ src, __bf16* __restrict__ dst, int n) {
  int i = blockIdx.x * blockDim.x + threadIdx.x;
  if (i < n) dst[i] = (__bf16)src[i];
}

// ------------------------------ gate / scale -------------------------------
__global__ void mask_kernel(const float* __restrict__ q, const float* __restrict__ mw,
                            const float* __restrict__ mb, float* __restrict__ scale) {
  int b = blockIdx.x;
  __shared__ float red[256];
  const float* qlast = q + ((size_t)b * SS + (SS - 1)) * DD;
  float s = 0.f;
  for (int i = threadIdx.x; i < DD; i += 256) s += qlast[i] * mw[i];
  red[threadIdx.x] = s;
  __syncthreads();
  for (int o = 128; o > 0; o >>= 1) {
    if (threadIdx.x < o) red[threadIdx.x] += red[threadIdx.x + o];
    __syncthreads();
  }
  if (threadIdx.x == 0) {
    float m = 1.f / (1.f + __expf(-(red[0] + mb[0])));
    scale[b] = (m < 0.1f) ? (1.f - m) : 1.f;
  }
}

// ------------------------------- WMMA GEMM ---------------------------------
// C[M,N] = (A[M,K](bf16) @ W[N,K](bf16)^T + bias[N]) * outScale.
// Block: 256 thr = 8 waves. Block tile 128M x 128N; wave = 16 rows x 128 cols
// (8 accumulators). W tile (128x32, 8KB) double-buffered in LDS via async DMA
// and shared by all 8 waves.  Grid: (N/128, M/128).
template <bool F32OUT>
__global__ __launch_bounds__(256) void gemm_kernel(const __bf16* __restrict__ A,
                                                   const __bf16* __restrict__ W,
                                                   const float* __restrict__ bias,
                                                   void* __restrict__ Cout,
                                                   int M, int N, int K, float outScale) {
  const int tid = threadIdx.x, lane = tid & 31, wid = tid >> 5;
  const int g = lane >> 4, ln = lane & 15;
  const int m0 = blockIdx.y * 128 + wid * 16;
  const int n0 = blockIdx.x * 128;

  __shared__ __bf16 ldsB[2][128][40];  // padded rows (80B) - no 2^k bank conflicts

  const __bf16* arow = A + (size_t)(m0 + ln) * K;
  // staging: 128 rows x 32 cols, 2 threads/row, 16 elements (2 x b128) each
  const int srow = tid >> 1, scol = (tid & 1) * 16;
  const __bf16* wsrc = W + (size_t)(n0 + srow) * K + scol;

  v8f acc[8] = {v8f{}, v8f{}, v8f{}, v8f{}, v8f{}, v8f{}, v8f{}, v8f{}};

  // prologue: stage k=0 into buffer 0
  copy_b128_to_lds(wsrc, &ldsB[0][srow][scol]);
  copy_b128_to_lds(wsrc + 8, &ldsB[0][srow][scol] + 8);

  for (int k = 0; k < K; k += 32) {
    const int cur = (k >> 5) & 1;
    const bool more = (k + 32) < K;
    if (more) {  // prefetch next W tile into the other buffer (overlaps compute)
      copy_b128_to_lds(wsrc + k + 32, &ldsB[cur ^ 1][srow][scol]);
      copy_b128_to_lds(wsrc + k + 40, &ldsB[cur ^ 1][srow][scol] + 8);
    }
    async_wait_upto(more ? 2 : 0);   // current tile done (async loads retire in order)
    __syncthreads();

    // A tile: lane = row m0+ln; K = g*16..g*16+15
    v8bf alo = *(const v8bf*)(arow + k + g * 16);
    v8bf ahi = *(const v8bf*)(arow + k + g * 16 + 8);
    v16bf va = __builtin_shufflevector(alo, ahi, 0, 1, 2, 3, 4, 5, 6, 7,
                                       8, 9, 10, 11, 12, 13, 14, 15);
#pragma unroll
    for (int t = 0; t < 8; ++t) {
      // B tile: lane = col (n-local) t*16+ln; K = g*16..g*16+15
      v16bf vb = load_bf16x16(&ldsB[cur][t * 16 + ln][g * 16]);
      acc[t] = WMMA_BF16(va, vb, acc[t]);
    }
    __syncthreads();   // all waves done reading before this buffer is re-staged
  }

#pragma unroll
  for (int t = 0; t < 8; ++t) {
    const int n = n0 + t * 16 + ln;
    const float bv = bias[n];
#pragma unroll
    for (int r = 0; r < 8; ++r) {
      const int m = m0 + r + 8 * g;
      const float v = (acc[t][r] + bv) * outScale;
      if (F32OUT)
        ((float*)Cout)[(size_t)m * N + n] = v;
      else
        ((__bf16*)Cout)[(size_t)m * N + n] = (__bf16)v;
    }
  }
}

// --------------------------- flash attention -------------------------------
// Grid: (S/QT, B*H). Block 256 thr = 8 waves, each wave owns 16 q rows.
// Qp (pre-scaled by 1/sqrt(dh)) / Kp / Vp laid out (B, S, H, DH) bf16.
// K/V^T tiles double-buffered; row-max via half-wave shuffles; row-sum via a
// fifth WMMA accumulator against an all-ones B operand (rowsum = P @ 1).
__global__ __launch_bounds__(256) void attn_kernel(const __bf16* __restrict__ Qp,
                                                   const __bf16* __restrict__ Kp,
                                                   const __bf16* __restrict__ Vp,
                                                   __bf16* __restrict__ AO) {
  const int tid = threadIdx.x, lane = tid & 31, wid = tid >> 5;
  const int g = lane >> 4, ln = lane & 15;
  const int b = blockIdx.y / HH, h = blockIdx.y % HH;

  __shared__ __bf16 ldsK[2][KT][72];   // K tiles (row=key), padded rows (144B)
  __shared__ __bf16 ldsVT[2][DH][40];  // V^T tiles (row=dh col), padded rows (80B)
  __shared__ __bf16 ldsP[8][16][40];   // per-wave probs (A-operand slab)

  const size_t headOff = (size_t)b * SS * DD + (size_t)h * DH;
  const __bf16* Qh = Qp + headOff;
  const __bf16* Kh = Kp + headOff;
  const __bf16* Vh = Vp + headOff;
  __bf16* Oh = AO + headOff;

  const int q0 = blockIdx.x * QT + wid * 16;

  // Preload this wave's Q A-tiles for the two dh-steps (dh 0..31, 32..63).
  const __bf16* qrow = Qh + (size_t)(q0 + ln) * DD;
  v16bf qa[2];
#pragma unroll
  for (int s = 0; s < 2; ++s) {
    v8bf lo = *(const v8bf*)(qrow + s * 32 + g * 16);
    v8bf hi = *(const v8bf*)(qrow + s * 32 + g * 16 + 8);
    qa[s] = __builtin_shufflevector(lo, hi, 0, 1, 2, 3, 4, 5, 6, 7,
                                    8, 9, 10, 11, 12, 13, 14, 15);
  }

  // constant all-ones B operand for the row-sum WMMA
  v16bf ones;
#pragma unroll
  for (int i = 0; i < 16; ++i) ones[i] = (__bf16)1.0f;

  v8f o[4] = {v8f{}, v8f{}, v8f{}, v8f{}};
  v8f osum = {};                      // running row sums (flash l), all N cols equal
  float rmax[8];
#pragma unroll
  for (int r = 0; r < 8; ++r) rmax[r] = -1e30f;

  // staging geometry: 32 rows x 64 cols, 8 threads/row, 8 elements each
  const int sr = tid >> 3, sc = (tid & 7) * 8;

  // prologue: async K tile 0, V tile 0 into registers
  copy_b128_to_lds(Kh + (size_t)sr * DD + sc, &ldsK[0][sr][sc]);
  v8bf vreg = *(const v8bf*)(Vh + (size_t)sr * DD + sc);

  for (int it = 0; it < SS / KT; ++it) {
    const int cur = it & 1;
    const bool more = (it + 1) < (SS / KT);

    // scatter current V tile (registers -> transposed LDS)
#pragma unroll
    for (int j = 0; j < 8; ++j) ldsVT[cur][sc + j][sr] = vreg[j];

    if (more) {  // prefetch next K (DMA) and next V (regs) - overlaps compute
      const size_t nxt = (size_t)((it + 1) * KT + sr) * DD + sc;
      copy_b128_to_lds(Kh + nxt, &ldsK[cur ^ 1][sr][sc]);
      vreg = *(const v8bf*)(Vh + nxt);
    }
    async_wait_upto(more ? 1 : 0);   // current K tile arrived
    __syncthreads();

    // ---- S = Q @ K^T : two 16x16 accumulators (key subtiles) ----
    v8f sacc[2];
#pragma unroll
    for (int t = 0; t < 2; ++t) {
      v8f c = {};
#pragma unroll
      for (int s = 0; s < 2; ++s) {
        v16bf bt = load_bf16x16(&ldsK[cur][t * 16 + ln][s * 32 + g * 16]);
        c = WMMA_BF16(qa[s], bt, c);
      }
      sacc[t] = c;
    }

    // ---- online softmax: max via shuffles, exp in regs, P -> LDS slab ----
#pragma unroll
    for (int r = 0; r < 8; ++r) {
      const float mx = group_max(fmaxf(sacc[0][r], sacc[1][r]));
      const float nm = fmaxf(rmax[r], mx);
      const float corr = __expf(rmax[r] - nm);
      rmax[r] = nm;
      const float p0 = __expf(sacc[0][r] - nm);
      const float p1 = __expf(sacc[1][r] - nm);
      ldsP[wid][r + 8 * g][ln] = (__bf16)p0;
      ldsP[wid][r + 8 * g][16 + ln] = (__bf16)p1;
      osum[r] *= corr;                 // running sum obeys same flash recurrence
#pragma unroll
      for (int t = 0; t < 4; ++t) o[t][r] *= corr;
    }

    // ---- O += P @ V ; rowsum += P @ 1 (all on the matrix pipe) ----
    v8bf plo = *(const v8bf*)(&ldsP[wid][ln][g * 16]);
    v8bf phi = *(const v8bf*)(&ldsP[wid][ln][g * 16 + 8]);
    v16bf pa = __builtin_shufflevector(plo, phi, 0, 1, 2, 3, 4, 5, 6, 7,
                                       8, 9, 10, 11, 12, 13, 14, 15);
#pragma unroll
    for (int t = 0; t < 4; ++t) {
      v16bf bt = load_bf16x16(&ldsVT[cur][t * 16 + ln][g * 16]);
      o[t] = WMMA_BF16(pa, bt, o[t]);
    }
    osum = WMMA_BF16(pa, ones, osum);

    __syncthreads();   // all waves done with this buffer pair before restaging
  }

  // ---- normalize and store (row r+8g's sum sits in osum[r], every column) ----
#pragma unroll
  for (int r = 0; r < 8; ++r) {
    const float inv = 1.f / osum[r];
    __bf16* orow = Oh + (size_t)(q0 + r + 8 * g) * DD;
#pragma unroll
    for (int t = 0; t < 4; ++t) orow[t * 16 + ln] = (__bf16)(o[t][r] * inv);
  }
}

// --------------- final broadcast: out[i,j,s,d] = f(m[i]) * full[j,s,d] -----
__global__ void bcast_kernel(const float* __restrict__ full, const float* __restrict__ scale,
                             float* __restrict__ out) {
  const size_t per = (size_t)BB * SS * DD;            // elements of full
  size_t idx = (size_t)blockIdx.x * blockDim.x + threadIdx.x;
  if (idx >= (size_t)BB * per) return;
  const int i = (int)(idx / per);
  out[idx] = scale[i] * full[idx - (size_t)i * per];
}

// ---------------------------------------------------------------------------
extern "C" void kernel_launch(void* const* d_in, const int* in_sizes, int n_in,
                              void* d_out, int out_size, void* d_ws, size_t ws_size,
                              hipStream_t stream) {
  const float* q   = (const float*)d_in[0];
  const float* k   = (const float*)d_in[1];
  const float* v   = (const float*)d_in[2];
  const float* inw = (const float*)d_in[3];   // (3D, D)
  const float* inb = (const float*)d_in[4];   // (3D)
  const float* ow  = (const float*)d_in[5];   // (D, D)
  const float* ob  = (const float*)d_in[6];   // (D)
  const float* mw  = (const float*)d_in[7];   // (1, D)
  const float* mb  = (const float*)d_in[8];   // (1)
  float* out = (float*)d_out;

  const size_t nQKV = (size_t)BB * SS * DD;   // 4 M elements
  const size_t nIW  = (size_t)3 * DD * DD;    // 3 M
  const size_t nOW  = (size_t)DD * DD;        // 1 M

  char* ws = (char*)d_ws;
  auto take = [&](size_t bytes) {
    void* p = (void*)ws;
    ws += (bytes + 255) & ~(size_t)255;
    return p;
  };
  __bf16* qb   = (__bf16*)take(nQKV * 2);
  __bf16* kb   = (__bf16*)take(nQKV * 2);
  __bf16* vb   = (__bf16*)take(nQKV * 2);
  __bf16* wib  = (__bf16*)take(nIW * 2);
  __bf16* wob  = (__bf16*)take(nOW * 2);
  __bf16* Qp   = (__bf16*)take(nQKV * 2);
  __bf16* Kp   = (__bf16*)take(nQKV * 2);
  __bf16* Vp   = (__bf16*)take(nQKV * 2);
  __bf16* AO   = (__bf16*)take(nQKV * 2);
  float*  FULL = (float*) take(nQKV * 4);
  float*  scl  = (float*) take(256);

  const int M = BB * SS;   // 4096

  // 1) convert to bf16
  cvt_kernel<<<(int)(nQKV / 256), 256, 0, stream>>>(q,   qb,  (int)nQKV);
  cvt_kernel<<<(int)(nQKV / 256), 256, 0, stream>>>(k,   kb,  (int)nQKV);
  cvt_kernel<<<(int)(nQKV / 256), 256, 0, stream>>>(v,   vb,  (int)nQKV);
  cvt_kernel<<<(int)(nIW  / 256), 256, 0, stream>>>(inw, wib, (int)nIW);
  cvt_kernel<<<(int)(nOW  / 256), 256, 0, stream>>>(ow,  wob, (int)nOW);

  // 2) gate scale per batch
  mask_kernel<<<BB, 256, 0, stream>>>(q, mw, mb, scl);

  // 3) QKV projections (bf16 out); Q pre-scaled by 1/sqrt(dh)
  dim3 ggrid(DD / 128, M / 128);
  gemm_kernel<false><<<ggrid, 256, 0, stream>>>(qb, wib,                       inb,          Qp, M, DD, DD, 0.125f);
  gemm_kernel<false><<<ggrid, 256, 0, stream>>>(kb, wib + (size_t)DD * DD,     inb + DD,     Kp, M, DD, DD, 1.0f);
  gemm_kernel<false><<<ggrid, 256, 0, stream>>>(vb, wib + (size_t)2 * DD * DD, inb + 2 * DD, Vp, M, DD, DD, 1.0f);

  // 4) flash attention
  attn_kernel<<<dim3(SS / QT, BB * HH), 256, 0, stream>>>(Qp, Kp, Vp, AO);

  // 5) output projection (f32 out)
  gemm_kernel<true><<<ggrid, 256, 0, stream>>>(AO, wob, ob, FULL, M, DD, DD, 1.0f);

  // 6) broadcast + gate scale into (B, B, S, D)
  const size_t total = (size_t)BB * BB * SS * DD;
  bcast_kernel<<<(int)((total + 255) / 256), 256, 0, stream>>>(FULL, scl, out);
}
